// AraBERTBiLSTMCRF_79602923864603
// MI455X (gfx1250) — matrix operand order repaired and checked by reading, not compiled, not run on hardware
//
#include <hip/hip_runtime.h>

typedef unsigned short ushort_t;
typedef __bf16 bf16x16 __attribute__((ext_vector_type(16)));
typedef __bf16 bf16x8  __attribute__((ext_vector_type(8)));
typedef float  v8f     __attribute__((ext_vector_type(8)));
typedef unsigned int v4u __attribute__((ext_vector_type(4)));
typedef int  v4i __attribute__((ext_vector_type(4)));
typedef int  v8i __attribute__((ext_vector_type(8)));

#define NB 32
#define NS 512
#define NE 768
#define NH 128
#define NG 512   /* 4*H */
#define NT 15

// ---------- helpers ----------

__device__ __forceinline__ ushort_t f2bf(float f) {
  unsigned u = __float_as_uint(f);
  u += 0x7fffu + ((u >> 16) & 1u);       // round-to-nearest-even
  return (ushort_t)(u >> 16);
}

__device__ __forceinline__ float sigmoidf_(float x) { return 1.0f / (1.0f + __expf(-x)); }

__device__ __forceinline__ bf16x16 cat16(bf16x8 lo, bf16x8 hi) {
  bf16x16 r;
#pragma unroll
  for (int i = 0; i < 8; ++i) { r[i] = lo[i]; r[i + 8] = hi[i]; }
  return r;
}

// A operand (16x32, bf16): lane<16 holds K {k0..k0+7, k0+16..k0+23} of row m=lane,
// lane>=16 holds K {k0+8..k0+15, k0+24..k0+31} of row m=lane-16.
__device__ __forceinline__ bf16x16 load_a(const ushort_t* base, int stride, int k0, int lane) {
  int m = lane & 15, sel = lane >> 4;
  const ushort_t* row = base + (size_t)m * stride + k0 + sel * 8;
  return cat16(*reinterpret_cast<const bf16x8*>(row),
               *reinterpret_cast<const bf16x8*>(row + 16));
}

// B operand (32x16, bf16), sourced from row-major W[n][k] (i.e. B = W^T):
// lane<16 holds K {k0..k0+7, k0+16..k0+23} of column n=lane, lane>=16 holds K {k0+8..k0+15, k0+24..k0+31}.
__device__ __forceinline__ bf16x16 load_b(const ushort_t* base, int stride, int k0, int lane) {
  int n = lane & 15, sel = lane >> 4;
  const ushort_t* row = base + (size_t)n * stride + k0 + sel * 8;
  return cat16(*reinterpret_cast<const bf16x8*>(row),
               *reinterpret_cast<const bf16x8*>(row + 16));
}

__device__ __forceinline__ v8f wmma_bf16(bf16x16 a, bf16x16 b, v8f c) {
  return __builtin_amdgcn_wmma_f32_16x16x32_bf16(false, a, false, b, (short)0, c, false, false);
}

// ---------- conversion kernels ----------

// x (B,S,E) f32 -> xbf (S*B, E) bf16 (fused transpose)
__global__ void xpose_bf16_kernel(const float* __restrict__ x, ushort_t* __restrict__ xbf) {
  size_t i = (size_t)blockIdx.x * blockDim.x + threadIdx.x;
  if (i >= (size_t)NB * NS * NE) return;
  size_t e = i % NE;
  size_t s = (i / NE) % NS;
  size_t b = i / ((size_t)NE * NS);
  xbf[(s * NB + b) * NE + e] = f2bf(x[i]);
}

__global__ void conv_bf16_kernel(const float* __restrict__ src, ushort_t* __restrict__ dst, int n) {
  int i = blockIdx.x * blockDim.x + threadIdx.x;
  if (i < n) dst[i] = f2bf(src[i]);
}

// Wp (15,256) -> Wpbf (16,256) bf16, row 15 zeroed
__global__ void wp_pad_kernel(const float* __restrict__ wp, ushort_t* __restrict__ dst) {
  int i = blockIdx.x * blockDim.x + threadIdx.x;
  if (i >= 16 * 256) return;
  int row = i >> 8, k = i & 255;
  dst[i] = (row < NT) ? f2bf(wp[row * 256 + k]) : (ushort_t)0;
}

// ---------- gate input GEMM:  Xg[dir][r][n] = xbf[r,:] . Wih[n,:] + bih[n]+bhh[n] ----------
// wave task = 32 rows x 64 cols, K = 768 (24 WMMA k-steps), 8 wmma / k-step

__global__ void gate_gemm_kernel(const ushort_t* __restrict__ xbf,
                                 const ushort_t* __restrict__ wih0,
                                 const ushort_t* __restrict__ wih1,
                                 const float* __restrict__ bih0, const float* __restrict__ bhh0,
                                 const float* __restrict__ bih1, const float* __restrict__ bhh1,
                                 float* __restrict__ xg) {
  int dir = blockIdx.y;
  const ushort_t* wih = dir ? wih1 : wih0;
  const float* bi = dir ? bih1 : bih0;
  const float* bh = dir ? bhh1 : bhh0;
  float* out = xg + (size_t)dir * ((size_t)NS * NB * NG);

  int wave = threadIdx.x >> 5;
  int lane = threadIdx.x & 31;
  int task = blockIdx.x * 8 + wave;   // 4096 wave tasks / dir
  int mt = task >> 3;                 // 512 row-tiles of 32
  int nt = task & 7;                  // 8 col-tiles of 64
  int row0 = mt * 32, col0 = nt * 64;

  v8f acc[2][4];
#pragma unroll
  for (int i = 0; i < 2; ++i)
#pragma unroll
    for (int j = 0; j < 4; ++j) acc[i][j] = (v8f)0.0f;

  const ushort_t* arow0 = xbf + (size_t)row0 * NE;
  const ushort_t* arow1 = xbf + (size_t)(row0 + 16) * NE;

  for (int k0 = 0; k0 < NE; k0 += 32) {
    __builtin_prefetch(arow0 + k0 + 256, 0, 0);   // global_prefetch_b8
    bf16x16 a0 = load_a(arow0, NE, k0, lane);
    bf16x16 a1 = load_a(arow1, NE, k0, lane);
#pragma unroll
    for (int j = 0; j < 4; ++j) {
      bf16x16 bm = load_b(wih + (size_t)(col0 + j * 16) * NE, NE, k0, lane);
      acc[0][j] = wmma_bf16(a0, bm, acc[0][j]);
      acc[1][j] = wmma_bf16(a1, bm, acc[1][j]);
    }
  }

  int nl = lane & 15;
  int selm = (lane >> 4) * 8;
#pragma unroll
  for (int j = 0; j < 4; ++j) {
    int n = col0 + j * 16 + nl;
    float bias = bi[n] + bh[n];
#pragma unroll
    for (int i = 0; i < 2; ++i)
#pragma unroll
      for (int r = 0; r < 8; ++r)
        out[(size_t)(row0 + i * 16 + selm + r) * NG + n] = acc[i][j][r] + bias;
  }
}

// ---------- persistent BiLSTM recurrence: one workgroup per direction ----------
// LDS: Whh bf16 (128KB) + h bf16 (8KB) + gate staging f32 (64KB)  = 200KB < 320KB/WGP
// Whh is staged by the Tensor Data Mover (tensor_load_to_lds): one wave issues the
// DMA descriptor, waits on TENSORcnt, everyone else waits at the workgroup barrier.

__global__ void lstm_kernel(const float* __restrict__ xg,
                            const ushort_t* __restrict__ whh0,
                            const ushort_t* __restrict__ whh1,
                            ushort_t* __restrict__ hbf0,
                            ushort_t* __restrict__ hbf1) {
  extern __shared__ char smem[];
  ushort_t* sWhh = (ushort_t*)smem;                                   // [NG][NH]
  ushort_t* sH   = (ushort_t*)(smem + (size_t)NG * NH * 2);           // [NB][NH]
  float*    sG   = (float*)   (smem + (size_t)NG * NH * 2 + NB * NH * 2); // [NB][NG]

  int dir = blockIdx.x;
  const float* xgd = xg + (size_t)dir * ((size_t)NS * NB * NG);
  const ushort_t* whh = dir ? whh1 : whh0;
  ushort_t* hout = dir ? hbf1 : hbf0;

  int tid = threadIdx.x;              // 512 threads = 16 waves
  int wave = tid >> 5, lane = tid & 31;

  // ---- TDM: DMA Whh (512x128 bf16 = 16384 x 8B units) into LDS ----
  if (wave == 0) {
    unsigned lds_off = (unsigned)(uintptr_t)(void*)sWhh;          // LDS byte address
    unsigned long long ga = (unsigned long long)(uintptr_t)whh;   // global byte address
    // D# group 0: count=1 | lds_addr | global_addr[56:0] | type=2
    v4u g0;
    g0.x = 1u;
    g0.y = lds_off;
    g0.z = (unsigned)(ga & 0xffffffffu);
    g0.w = (unsigned)((ga >> 32) & 0x01ffffffu) | 0x80000000u;
    // D# group 1: data_size=3 (8B); tensor_dim0=tile_dim0=0x4000; dim1=tile1=1;
    // stride0=stride1=0x4000 units
    v8i g1;
    g1[0] = (int)0x00030000u;   // wg_mask=0, data_size=3
    g1[1] = (int)0x40000000u;   // tensor_dim0[15:0]=0x4000 in bits[63:48]
    g1[2] = (int)0x00010000u;   // tensor_dim0 hi=0; tensor_dim1 lo=1
    g1[3] = (int)0x40000000u;   // tensor_dim1 hi=0; tile_dim0=0x4000
    g1[4] = 0x00000001;         // tile_dim1=1, tile_dim2=0
    g1[5] = 0x00004000;         // tensor_dim0_stride lo32 = 0x4000
    g1[6] = (int)0x40000000u;   // stride0 hi=0; tensor_dim1_stride lo16=0x4000
    g1[7] = 0;                  // tensor_dim1_stride hi=0
    v4i gz4 = { 0, 0, 0, 0 };
    v8i gz8 = { 0, 0, 0, 0, 0, 0, 0, 0 };
    __builtin_amdgcn_tensor_load_to_lds(g0, g1, gz4, gz4, gz8, 0);
    __builtin_amdgcn_s_wait_tensorcnt(0);
  }
  for (int i = tid; i < NB * NH; i += 512) sH[i] = 0;

  float cst[8];
#pragma unroll
  for (int j = 0; j < 8; ++j) cst[j] = 0.0f;
  __syncthreads();

  int nl = lane & 15;
  int selm = (lane >> 4) * 8;

  for (int s = 0; s < NS; ++s) {
    int t = dir ? (NS - 1 - s) : s;
    const float* xrow = xgd + (size_t)t * NB * NG;

    // ---- G = Xg[t] + h @ Whh^T  (wave: 2M x 2N tiles, K=128) ----
    v8f acc[2][2];
#pragma unroll
    for (int mi = 0; mi < 2; ++mi)
#pragma unroll
      for (int nj = 0; nj < 2; ++nj) {
        int nc = wave * 32 + nj * 16 + nl;
#pragma unroll
        for (int r = 0; r < 8; ++r)
          acc[mi][nj][r] = xrow[(size_t)(mi * 16 + selm + r) * NG + nc];
      }

#pragma unroll
    for (int k0 = 0; k0 < NH; k0 += 32) {
      bf16x16 a0 = load_a(sH, NH, k0, lane);
      bf16x16 a1 = load_a(sH + 16 * NH, NH, k0, lane);
      bf16x16 b0 = load_b(sWhh + (size_t)(wave * 32) * NH, NH, k0, lane);
      bf16x16 b1 = load_b(sWhh + (size_t)(wave * 32 + 16) * NH, NH, k0, lane);
      acc[0][0] = wmma_bf16(a0, b0, acc[0][0]);
      acc[0][1] = wmma_bf16(a0, b1, acc[0][1]);
      acc[1][0] = wmma_bf16(a1, b0, acc[1][0]);
      acc[1][1] = wmma_bf16(a1, b1, acc[1][1]);
    }

#pragma unroll
    for (int mi = 0; mi < 2; ++mi)
#pragma unroll
      for (int nj = 0; nj < 2; ++nj) {
        int nc = wave * 32 + nj * 16 + nl;
#pragma unroll
        for (int r = 0; r < 8; ++r)
          sG[(size_t)(mi * 16 + selm + r) * NG + nc] = acc[mi][nj][r];
      }
    __syncthreads();

    // ---- elementwise LSTM cell: each thread owns 8 (b,hu) pairs ----
#pragma unroll
    for (int j = 0; j < 8; ++j) {
      int bh = tid + j * 512;
      int b = bh >> 7, hu = bh & 127;
      float zi = sG[b * NG + hu];
      float zf = sG[b * NG + NH + hu];
      float zg = sG[b * NG + 2 * NH + hu];
      float zo = sG[b * NG + 3 * NH + hu];
      float c = sigmoidf_(zf) * cst[j] + sigmoidf_(zi) * tanhf(zg);
      cst[j] = c;
      float h = sigmoidf_(zo) * tanhf(c);
      ushort_t hb = f2bf(h);
      sH[b * NH + hu] = hb;
      hout[((size_t)t * NB + b) * NH + hu] = hb;
    }
    __syncthreads();
  }
}

// ---------- emissions: em[r][j] = [hf|hb][r,:] . Wp[j,:] + bp[j], N padded to 16 ----------

__global__ void emis_kernel(const ushort_t* __restrict__ hf,
                            const ushort_t* __restrict__ hb,
                            const ushort_t* __restrict__ wpbf,
                            const float* __restrict__ bp,
                            float* __restrict__ em) {
  int wave = threadIdx.x >> 5, lane = threadIdx.x & 31;
  int task = blockIdx.x * 8 + wave;       // 1024 tasks of 16 rows
  int row0 = task * 16;
  int m = lane & 15, sel = lane >> 4;

  v8f acc = (v8f)0.0f;
#pragma unroll
  for (int k0 = 0; k0 < 2 * NH; k0 += 32) {
    int klo = k0 + sel * 8;
    int khi = k0 + 16 + sel * 8;
    const ushort_t* pl = (klo < NH) ? hf + (size_t)(row0 + m) * NH + klo
                                    : hb + (size_t)(row0 + m) * NH + (klo - NH);
    const ushort_t* ph = (khi < NH) ? hf + (size_t)(row0 + m) * NH + khi
                                    : hb + (size_t)(row0 + m) * NH + (khi - NH);
    bf16x16 a = cat16(*reinterpret_cast<const bf16x8*>(pl),
                      *reinterpret_cast<const bf16x8*>(ph));
    bf16x16 b = load_b(wpbf, 2 * NH, k0, lane);
    acc = wmma_bf16(a, b, acc);
  }

  int n = lane & 15;
  int selm = (lane >> 4) * 8;
  float bias = (n < NT) ? bp[n] : 0.0f;
#pragma unroll
  for (int r = 0; r < 8; ++r)
    em[(size_t)(row0 + selm + r) * 16 + n] = acc[r] + bias;
}

// ---------- CRF NLL: one wave32, one batch element per lane ----------

__global__ void crf_kernel(const float* __restrict__ em,      // [S][B][16]
                           const int* __restrict__ tags,      // [B][S]
                           const int* __restrict__ mask,      // [B][S]
                           const float* __restrict__ trans,
                           const float* __restrict__ start_t,
                           const float* __restrict__ end_t,
                           float* __restrict__ out) {
  __shared__ float sT[NT * NT], sS[NT], sE[NT];
  int lane = threadIdx.x;
  for (int i = lane; i < NT * NT; i += 32) sT[i] = trans[i];
  if (lane < NT) { sS[lane] = start_t[lane]; sE[lane] = end_t[lane]; }
  __syncthreads();

  int b = lane;
  const int* tgb = tags + b * NS;
  const int* mkb = mask + b * NS;

  int prev = tgb[0];
  float score = sS[prev] + em[(size_t)b * 16 + prev];
  int seqlen = mkb[0];

  float alpha[NT];
#pragma unroll
  for (int j = 0; j < NT; ++j) alpha[j] = sS[j] + em[(size_t)b * 16 + j];

  for (int s = 1; s < NS; ++s) {
    const float* emrow = em + ((size_t)s * NB + b) * 16;
    int tg = tgb[s];
    int mk = mkb[s];
    score += (sT[prev * NT + tg] + emrow[tg]) * (float)mk;
    prev = tg;
    seqlen += mk;
    if (mk > 0) {
      float nxt[NT];
#pragma unroll
      for (int j2 = 0; j2 < NT; ++j2) {
        float mx = -1e30f;
#pragma unroll
        for (int j = 0; j < NT; ++j) mx = fmaxf(mx, alpha[j] + sT[j * NT + j2]);
        float sum = 0.0f;
#pragma unroll
        for (int j = 0; j < NT; ++j) sum += __expf(alpha[j] + sT[j * NT + j2] - mx);
        nxt[j2] = mx + __logf(sum) + emrow[j2];
      }
#pragma unroll
      for (int j2 = 0; j2 < NT; ++j2) alpha[j2] = nxt[j2];
    }
  }

  score += sE[tgb[seqlen - 1]];

  float mx = -1e30f;
#pragma unroll
  for (int j = 0; j < NT; ++j) mx = fmaxf(mx, alpha[j] + sE[j]);
  float sum = 0.0f;
#pragma unroll
  for (int j = 0; j < NT; ++j) sum += __expf(alpha[j] + sE[j] - mx);
  float logZ = mx + __logf(sum);

  float nll = -(score - logZ);
#pragma unroll
  for (int off = 16; off > 0; off >>= 1) nll += __shfl_down(nll, off, 32);
  if (lane == 0) out[0] = nll;
}

// ---------- launch ----------

extern "C" void kernel_launch(void* const* d_in, const int* in_sizes, int n_in,
                              void* d_out, int out_size, void* d_ws, size_t ws_size,
                              hipStream_t stream) {
  (void)in_sizes; (void)n_in; (void)out_size; (void)ws_size;

  const float* x      = (const float*)d_in[0];
  const int*   tags   = (const int*)d_in[1];
  const int*   mask   = (const int*)d_in[2];
  const float* Wih_f  = (const float*)d_in[3];
  const float* Whh_f  = (const float*)d_in[4];
  const float* bih_f  = (const float*)d_in[5];
  const float* bhh_f  = (const float*)d_in[6];
  const float* Wih_b  = (const float*)d_in[7];
  const float* Whh_b  = (const float*)d_in[8];
  const float* bih_b  = (const float*)d_in[9];
  const float* bhh_b  = (const float*)d_in[10];
  const float* Wp     = (const float*)d_in[11];
  const float* bp     = (const float*)d_in[12];
  const float* trans  = (const float*)d_in[13];
  const float* start_t = (const float*)d_in[14];
  const float* end_t   = (const float*)d_in[15];

  char* ws = (char*)d_ws;
  size_t off = 0;
  auto wsalloc = [&](size_t bytes) -> void* {
    void* p = ws + off;
    off += (bytes + 255) & ~(size_t)255;
    return p;
  };

  const size_t M = (size_t)NS * NB;   // 16384
  ushort_t* xbf   = (ushort_t*)wsalloc(M * NE * 2);
  ushort_t* wihf  = (ushort_t*)wsalloc((size_t)NG * NE * 2);
  ushort_t* wihb  = (ushort_t*)wsalloc((size_t)NG * NE * 2);
  ushort_t* whhf  = (ushort_t*)wsalloc((size_t)NG * NH * 2);
  ushort_t* whhb  = (ushort_t*)wsalloc((size_t)NG * NH * 2);
  ushort_t* wpbf  = (ushort_t*)wsalloc((size_t)16 * 256 * 2);
  float*    xg    = (float*)wsalloc(2 * M * NG * 4);
  ushort_t* hbf_f = (ushort_t*)wsalloc(M * NH * 2);
  ushort_t* hbf_b = (ushort_t*)wsalloc(M * NH * 2);
  float*    em    = (float*)wsalloc(M * 16 * 4);

  // 1) precision conversion / layout
  xpose_bf16_kernel<<<(unsigned)((M * NE + 255) / 256), 256, 0, stream>>>(x, xbf);
  conv_bf16_kernel<<<(NG * NE + 255) / 256, 256, 0, stream>>>(Wih_f, wihf, NG * NE);
  conv_bf16_kernel<<<(NG * NE + 255) / 256, 256, 0, stream>>>(Wih_b, wihb, NG * NE);
  conv_bf16_kernel<<<(NG * NH + 255) / 256, 256, 0, stream>>>(Whh_f, whhf, NG * NH);
  conv_bf16_kernel<<<(NG * NH + 255) / 256, 256, 0, stream>>>(Whh_b, whhb, NG * NH);
  wp_pad_kernel<<<16, 256, 0, stream>>>(Wp, wpbf);

  // 2) big input GEMM (both directions), bias folded in
  gate_gemm_kernel<<<dim3(512, 2), 256, 0, stream>>>(xbf, wihf, wihb,
                                                     bih_f, bhh_f, bih_b, bhh_b, xg);

  // 3) persistent recurrence: Whh DMA'd into 320KB LDS by the TDM, one WGP per direction
  size_t lds_bytes = (size_t)NG * NH * 2 + (size_t)NB * NH * 2 + (size_t)NB * NG * 4;
  lstm_kernel<<<2, 512, lds_bytes, stream>>>(xg, whhf, whhb, hbf_f, hbf_b);

  // 4) emission projection (WMMA, T padded to 16)
  emis_kernel<<<128, 256, 0, stream>>>(hbf_f, hbf_b, wpbf, bp, em);

  // 5) CRF NLL, one wave32 (lane = batch element), scalar out
  crf_kernel<<<1, 32, 0, stream>>>(em, tags, mask, trans, start_t, end_t, (float*)d_out);
}
